// DecoderLM_55207509623388
// MI455X (gfx1250) — compile-verified, matrix operand order checked by reading, not run on hardware
//
#include <hip/hip_runtime.h>
#include <hip/hip_bf16.h>
#include <math.h>

// Problem constants (match reference)
#define Lc   6
#define Dc   1024
#define DFFc 4096
#define Vc   32000
#define Sc   1024
#define Bc   2
#define Hc   16
#define DKc  64
#define Mc   (Bc * Sc)        // 2048 rows of activations
#define NEG_BIG (-1e30f)

typedef __attribute__((ext_vector_type(16))) __bf16        v16bf;
typedef __attribute__((ext_vector_type(2)))  __bf16        v2bf;
typedef __attribute__((ext_vector_type(8)))  float         v8f;
typedef __attribute__((ext_vector_type(8)))  unsigned int  v8u;

// K index (even, pair base) for fragment VGPR v (0..7), lane-half lh (0/1).
// Per CDNA5 ISA 7.12.2: 16-bit A 16x32 — VGPR0..3 hold K=0..7 (+8 for lanes 16..31),
// VGPR4..7 hold K=16..23 (+8 for lanes 16..31). B mirrors with N in lane.
__device__ __forceinline__ int frag_k(int v, int lh) {
    int base = (v < 4) ? (v * 2) : (16 + (v - 4) * 2);
    return base + lh * 8;
}

// Pack two f32 -> packed bf16 pair (lo = a, hi = b) in exactly one VALU op.
// Inline asm pins the instruction so SLP can't pair adjacent packs into
// wrong-element cvt_pk + half-move repair sequences.
__device__ __forceinline__ unsigned pack_bf16(float a, float b) {
#if defined(__gfx1250__)
    unsigned r;
    asm("v_cvt_pk_bf16_f32 %0, %1, %2" : "=v"(r) : "v"(a), "v"(b));
    return r;
#elif __has_builtin(__builtin_amdgcn_cvt_pk_bf16_f32)
    v2bf p = __builtin_amdgcn_cvt_pk_bf16_f32(a, b);
    return __builtin_bit_cast(unsigned, p);
#else
    unsigned short lo = __builtin_bit_cast(unsigned short, (__bf16)a);
    unsigned short hi = __builtin_bit_cast(unsigned short, (__bf16)b);
    return ((unsigned)hi << 16) | (unsigned)lo;
#endif
}

// Load one 16x32 fragment whose K-pairs live contiguously in a (4B-aligned)
// row of LDS: 8 x ds_load_b32 (merged to ds_load_2addr_b32), no repacking.
__device__ __forceinline__ v16bf load_frag_row(const __bf16* rowPtr, int lh) {
    const unsigned int* r32 = (const unsigned int*)rowPtr;
    v8u u;
#pragma unroll
    for (int v = 0; v < 8; ++v) u[v] = r32[frag_k(v, lh) >> 1];
    return __builtin_bit_cast(v16bf, u);
}

__device__ __forceinline__ v8f wmma_bf16(v16bf a, v16bf b, v8f c) {
    return __builtin_amdgcn_wmma_f32_16x16x32_bf16(
        /*neg_a=*/false, a, /*neg_b=*/false, b,
        /*c_mod=*/(short)0, c, /*reuse_a=*/false, /*reuse_b=*/false);
}

__device__ __forceinline__ float gelu_exact(float x) {
    return 0.5f * x * (1.0f + erff(x * 0.70710678118654752f));
}

// ---------------------------------------------------------------------------
// Embedding: h[b,s,:] = tok_emb[x[b,s],:] + pos_emb[s,:]
// ---------------------------------------------------------------------------
__global__ __launch_bounds__(256) void embed_k(
    const int* __restrict__ x, const float* __restrict__ tok,
    const float* __restrict__ pos, float* __restrict__ h, int total)
{
    int i = blockIdx.x * 256 + threadIdx.x;
    if (i >= total) return;
    int d  = i % Dc;
    int bs = i / Dc;
    int s  = bs % Sc;
    h[i] = tok[(size_t)x[bs] * Dc + d] + pos[(size_t)s * Dc + d];
}

// ---------------------------------------------------------------------------
// LayerNorm over last dim (Dn), one block (8 waves) per row.
// ---------------------------------------------------------------------------
__global__ __launch_bounds__(256) void layernorm_k(
    const float* __restrict__ x, const float* __restrict__ g,
    const float* __restrict__ b, float* __restrict__ y, int Dn)
{
    __shared__ float red[8];
    __shared__ float sMean, sRstd;
    const int row = blockIdx.x;
    const float* xr = x + (size_t)row * Dn;

    float s = 0.f;
    for (int i = threadIdx.x; i < Dn; i += 256) s += xr[i];
#pragma unroll
    for (int mm = 16; mm >= 1; mm >>= 1) s += __shfl_xor(s, mm, 32);
    if ((threadIdx.x & 31) == 0) red[threadIdx.x >> 5] = s;
    __syncthreads();
    if (threadIdx.x == 0) {
        float t = 0.f;
#pragma unroll
        for (int j = 0; j < 8; ++j) t += red[j];
        sMean = t / (float)Dn;
    }
    __syncthreads();
    const float mean = sMean;

    float vv = 0.f;
    for (int i = threadIdx.x; i < Dn; i += 256) {
        float d = xr[i] - mean; vv += d * d;
    }
#pragma unroll
    for (int mm = 16; mm >= 1; mm >>= 1) vv += __shfl_xor(vv, mm, 32);
    __syncthreads();                 // red[] free again
    if ((threadIdx.x & 31) == 0) red[threadIdx.x >> 5] = vv;
    __syncthreads();
    if (threadIdx.x == 0) {
        float t = 0.f;
#pragma unroll
        for (int j = 0; j < 8; ++j) t += red[j];
        sRstd = rsqrtf(t / (float)Dn + 1e-5f);
    }
    __syncthreads();
    const float rstd = sRstd;

    for (int i = threadIdx.x; i < Dn; i += 256)
        y[(size_t)row * Dn + i] = g[i] * (xr[i] - mean) * rstd + b[i];
}

// ---------------------------------------------------------------------------
// C[M,N] = epilogue(A[M,K] @ W[K,N] + bias), f32 in/out, bf16 WMMA compute.
// Block: 256 threads = 8 waves. Tile 128(M) x 128(N), K-step 32, double-
// buffered LDS (stage k+1 while computing k; one barrier per K-step).
// Wave w: rows {(w&3)*16, (w&3)*16+64}, cols (w>>2)*64 -> 8 WMMAs per K-step
// from 2 A-fragments and 4 B-fragments (all loaded before the WMMA burst to
// avoid WMMA->VALU WAR hazard NOPs).
// Staging issues ALL global float4 loads into distinct registers first (so
// they clause and overlap), then packs with v_cvt_pk_bf16_f32 and stores.
// A staged row-major (packed K-pairs), B staged TRANSPOSED ([n][k]) so both
// fragment types read packed 32-bit K-pairs from a single LDS row.
// Row stride 38 halves = 76B: 4B-aligned and odd word stride (no bank conflicts).
// mode 0: +bias ; mode 1: +bias+residual ; mode 2: gelu(+bias)
// ---------------------------------------------------------------------------
#define BMt 128
#define BNt 128
#define BKt 32

__global__ __launch_bounds__(256) void gemm_bf16_wmma(
    const float* __restrict__ A, const float* __restrict__ W,
    const float* __restrict__ bias, const float* __restrict__ residual,
    float* __restrict__ C, int M, int N, int K, int mode)
{
    __shared__ __bf16 As[2][BMt][38];
    __shared__ __bf16 Bp[2][BNt][38];   // transposed: [n][k]

    const int tid  = threadIdx.x;
    const int lane = tid & 31;
    const int wave = tid >> 5;
    const int lh   = lane >> 4;
    const int l16  = lane & 15;
    const int wm   = wave & 3;       // row group (16 rows, x2 with +64)
    const int wn   = wave >> 2;      // col group of 64
    const int m0   = blockIdx.y * BMt;
    const int n0   = blockIdx.x * BNt;

    v8f acc[2][4] = {};

    // staging assignments
    const int arow = tid >> 1;            // 0..127
    const int akb  = (tid & 1) * 16;      // 0 or 16 (8 pairs each)
    const int bn4  = (tid & 31) * 4;      // n chunk of 4 (0..124)
    const int bkg  = tid >> 5;            // 0..7 pair-group

    auto stage = [&](int buf, int k0) {
        // ---- issue ALL global loads first (distinct regs -> clause + overlap)
        const float4* apg4 = (const float4*)(A + (size_t)(m0 + arow) * K + (k0 + akb));
        float4 fa[4];
#pragma unroll
        for (int j = 0; j < 4; ++j) fa[j] = apg4[j];
        float4 fb[2][2];
#pragma unroll
        for (int it = 0; it < 2; ++it) {
            const int kpi = bkg + 8 * it;               // pair index 0..15
            fb[it][0] = *(const float4*)(W + (size_t)(k0 + 2*kpi)     * N + (n0 + bn4));
            fb[it][1] = *(const float4*)(W + (size_t)(k0 + 2*kpi + 1) * N + (n0 + bn4));
        }
        // ---- pack + store A (16 consecutive floats -> 8 packed pairs)
        unsigned* dstA = (unsigned*)&As[buf][arow][0];
#pragma unroll
        for (int j = 0; j < 4; ++j) {
            dstA[(akb >> 1) + 2*j + 0] = pack_bf16(fa[j].x, fa[j].y);
            dstA[(akb >> 1) + 2*j + 1] = pack_bf16(fa[j].z, fa[j].w);
        }
        // ---- pack + store B transposed (pairs along K at fixed n)
#pragma unroll
        for (int it = 0; it < 2; ++it) {
            const int kpi = bkg + 8 * it;
            ((unsigned*)&Bp[buf][bn4 + 0][0])[kpi] = pack_bf16(fb[it][0].x, fb[it][1].x);
            ((unsigned*)&Bp[buf][bn4 + 1][0])[kpi] = pack_bf16(fb[it][0].y, fb[it][1].y);
            ((unsigned*)&Bp[buf][bn4 + 2][0])[kpi] = pack_bf16(fb[it][0].z, fb[it][1].z);
            ((unsigned*)&Bp[buf][bn4 + 3][0])[kpi] = pack_bf16(fb[it][0].w, fb[it][1].w);
        }
    };

    stage(0, 0);
    __syncthreads();
    int cur = 0;
    for (int k0 = 0; k0 < K; k0 += BKt) {
        if (k0 + BKt < K) stage(cur ^ 1, k0 + BKt);

        const v16bf a0 = load_frag_row(&As[cur][wm * 16 + l16][0], lh);
        const v16bf a1 = load_frag_row(&As[cur][wm * 16 + 64 + l16][0], lh);
        const v16bf b0 = load_frag_row(&Bp[cur][wn * 64 +  0 + l16][0], lh);
        const v16bf b1 = load_frag_row(&Bp[cur][wn * 64 + 16 + l16][0], lh);
        const v16bf b2 = load_frag_row(&Bp[cur][wn * 64 + 32 + l16][0], lh);
        const v16bf b3 = load_frag_row(&Bp[cur][wn * 64 + 48 + l16][0], lh);

        acc[0][0] = wmma_bf16(a0, b0, acc[0][0]);
        acc[0][1] = wmma_bf16(a0, b1, acc[0][1]);
        acc[0][2] = wmma_bf16(a0, b2, acc[0][2]);
        acc[0][3] = wmma_bf16(a0, b3, acc[0][3]);
        acc[1][0] = wmma_bf16(a1, b0, acc[1][0]);
        acc[1][1] = wmma_bf16(a1, b1, acc[1][1]);
        acc[1][2] = wmma_bf16(a1, b2, acc[1][2]);
        acc[1][3] = wmma_bf16(a1, b3, acc[1][3]);

        __syncthreads();
        cur ^= 1;
    }

#pragma unroll
    for (int r = 0; r < 2; ++r) {
#pragma unroll
        for (int t2 = 0; t2 < 4; ++t2) {
#pragma unroll
            for (int v = 0; v < 8; ++v) {
                const int row = m0 + wm * 16 + r * 64 + v + lh * 8;
                const int col = n0 + wn * 64 + t2 * 16 + l16;
                float x0 = acc[r][t2][v] + bias[col];
                if (mode == 1)      x0 += residual[(size_t)row * N + col];
                else if (mode == 2) x0 = gelu_exact(x0);
                C[(size_t)row * N + col] = x0;
            }
        }
    }
}

// ---------------------------------------------------------------------------
// Fused causal attention, flash-style. Q/K/V/O: [B*S, D] with head h at
// columns [h*64, h*64+64). Grid (B*H, S/128), 256 threads = 8 waves,
// wave handles 16 query rows; streams 32-key blocks with online softmax.
// ---------------------------------------------------------------------------
__global__ __launch_bounds__(256) void attention_k(
    const float* __restrict__ Q, const float* __restrict__ Kb,
    const float* __restrict__ Vb, float* __restrict__ O)
{
    __shared__ __bf16 Pst[8][16][38];     // per-wave P staging

    const int bh   = blockIdx.x;
    const int b    = bh / Hc;
    const int hh   = bh % Hc;
    const int wave = threadIdx.x >> 5;
    const int lane = threadIdx.x & 31;
    const int lh   = lane >> 4;
    const int l16  = lane & 15;
    const int q0   = blockIdx.y * 128 + wave * 16;
    const size_t base = ((size_t)b * Sc) * Dc + (size_t)hh * DKc;

    // Q fragments for dk 0..31 and 32..63 (A layout: M=row in lane, K packed)
    v16bf aq0, aq1;
    {
        const float2* qr2 = (const float2*)(Q + base + (size_t)(q0 + l16) * Dc);
        v8u u0, u1;
#pragma unroll
        for (int v = 0; v < 8; ++v) {
            const int kp = frag_k(v, lh) >> 1;
            const float2 p0 = qr2[kp];
            const float2 p1 = qr2[16 + kp];
            u0[v] = pack_bf16(p0.x, p0.y);
            u1[v] = pack_bf16(p1.x, p1.y);
        }
        aq0 = __builtin_bit_cast(v16bf, u0);
        aq1 = __builtin_bit_cast(v16bf, u1);
    }

    float m_i[8], l_i[8];
    v8f accO[4] = {};
#pragma unroll
    for (int v = 0; v < 8; ++v) { m_i[v] = NEG_BIG; l_i[v] = 0.f; }

    const float scale = 0.125f;   // 1/sqrt(64)

    for (int kbk = 0; kbk <= q0 + 15; kbk += 32) {
        // ----- K fragments for both 16-key halves, loaded up front -----
        v8u uk[2][2];
#pragma unroll
        for (int t = 0; t < 2; ++t) {
            const int key = kbk + t * 16 + l16;      // column of B fragment
            const float2* kr2 = (const float2*)(Kb + base + (size_t)key * Dc);
#pragma unroll
            for (int v = 0; v < 8; ++v) {
                const int kp = frag_k(v, lh) >> 1;   // contraction dim = dk pairs
                const float2 p0 = kr2[kp];
                const float2 p1 = kr2[16 + kp];
                uk[t][0][v] = pack_bf16(p0.x, p0.y);
                uk[t][1][v] = pack_bf16(p1.x, p1.y);
            }
        }
        // ----- scores: 16 queries x 32 keys (4 WMMAs, no interleaved loads) --
        v8f sc[2] = {};
        sc[0] = wmma_bf16(aq0, __builtin_bit_cast(v16bf, uk[0][0]), sc[0]);
        sc[1] = wmma_bf16(aq0, __builtin_bit_cast(v16bf, uk[1][0]), sc[1]);
        sc[0] = wmma_bf16(aq1, __builtin_bit_cast(v16bf, uk[0][1]), sc[0]);
        sc[1] = wmma_bf16(aq1, __builtin_bit_cast(v16bf, uk[1][1]), sc[1]);

        // scale + causal mask
#pragma unroll
        for (int t = 0; t < 2; ++t) {
            const int key = kbk + t * 16 + l16;
#pragma unroll
            for (int v = 0; v < 8; ++v) {
                const int qrow = q0 + v + lh * 8;
                float s = sc[t][v] * scale;
                sc[t][v] = (key > qrow) ? NEG_BIG : s;
            }
        }
        // online softmax (row reductions across 16-lane halves)
#pragma unroll
        for (int v = 0; v < 8; ++v) {
            float rm = fmaxf(sc[0][v], sc[1][v]);
#pragma unroll
            for (int mm = 1; mm <= 8; mm <<= 1) rm = fmaxf(rm, __shfl_xor(rm, mm, 32));
            const float mnew = fmaxf(m_i[v], rm);
            const float sold = __expf(m_i[v] - mnew);
            const float p0 = __expf(sc[0][v] - mnew);
            const float p1 = __expf(sc[1][v] - mnew);
            sc[0][v] = p0; sc[1][v] = p1;
            float rs = p0 + p1;
#pragma unroll
            for (int mm = 1; mm <= 8; mm <<= 1) rs += __shfl_xor(rs, mm, 32);
            l_i[v] = l_i[v] * sold + rs;
            m_i[v] = mnew;
#pragma unroll
            for (int t2 = 0; t2 < 4; ++t2) accO[t2][v] = accO[t2][v] * sold;
        }
        // ----- stage P (C layout -> LDS) and re-read as A fragment -----
#pragma unroll
        for (int t = 0; t < 2; ++t)
#pragma unroll
            for (int v = 0; v < 8; ++v)
                Pst[wave][v + lh * 8][t * 16 + l16] = (__bf16)sc[t][v];
        asm volatile("s_wait_dscnt 0x0" ::: "memory");
        const v16bf apf = load_frag_row(&Pst[wave][l16][0], lh);
        // ----- V fragments for all 4 dk-tiles, then the 4 WMMAs -----
        v8u uv[4];
#pragma unroll
        for (int t2 = 0; t2 < 4; ++t2) {
#pragma unroll
            for (int v = 0; v < 8; ++v) {
                const int kk = frag_k(v, lh);        // key within block
                const float* vr = Vb + base + (size_t)(kbk + kk) * Dc + t2 * 16 + l16;
                uv[t2][v] = pack_bf16(vr[0], vr[Dc]);
            }
        }
        accO[0] = wmma_bf16(apf, __builtin_bit_cast(v16bf, uv[0]), accO[0]);
        accO[1] = wmma_bf16(apf, __builtin_bit_cast(v16bf, uv[1]), accO[1]);
        accO[2] = wmma_bf16(apf, __builtin_bit_cast(v16bf, uv[2]), accO[2]);
        accO[3] = wmma_bf16(apf, __builtin_bit_cast(v16bf, uv[3]), accO[3]);
    }

    // normalize + write out
#pragma unroll
    for (int t2 = 0; t2 < 4; ++t2)
#pragma unroll
        for (int v = 0; v < 8; ++v) {
            const int qrow = q0 + v + lh * 8;
            O[base + (size_t)qrow * Dc + t2 * 16 + l16] = accO[t2][v] / l_i[v];
        }
}

// ---------------------------------------------------------------------------
extern "C" void kernel_launch(void* const* d_in, const int* in_sizes, int n_in,
                              void* d_out, int out_size, void* d_ws, size_t ws_size,
                              hipStream_t stream) {
    (void)in_sizes; (void)n_in; (void)out_size; (void)ws_size;

    const int*   x       = (const int*)  d_in[0];
    const float* tok_emb = (const float*)d_in[1];
    const float* pos_emb = (const float*)d_in[2];
    const float* ln1_g   = (const float*)d_in[3];
    const float* ln1_b   = (const float*)d_in[4];
    const float* Wq      = (const float*)d_in[5];
    const float* bq      = (const float*)d_in[6];
    const float* Wk      = (const float*)d_in[7];
    const float* bk      = (const float*)d_in[8];
    const float* Wv      = (const float*)d_in[9];
    const float* bv      = (const float*)d_in[10];
    const float* Wo      = (const float*)d_in[11];
    const float* bo      = (const float*)d_in[12];
    const float* ln2_g   = (const float*)d_in[13];
    const float* ln2_b   = (const float*)d_in[14];
    const float* W1      = (const float*)d_in[15];
    const float* b1      = (const float*)d_in[16];
    const float* W2      = (const float*)d_in[17];
    const float* b2      = (const float*)d_in[18];
    const float* lnf_g   = (const float*)d_in[19];
    const float* lnf_b   = (const float*)d_in[20];
    const float* Wh      = (const float*)d_in[21];
    const float* bh      = (const float*)d_in[22];
    float* out = (float*)d_out;

    // Workspace layout (floats): [h: M*D][y: M*D][big: M*DFF]
    //   big serves as q/k/v (3 * M*D <= M*DFF) during attention,
    //   then as the FFN hidden activation. Total ~50 MB.
    float* ws  = (float*)d_ws;
    const size_t MD = (size_t)Mc * Dc;
    float* h   = ws;
    float* y   = ws + MD;
    float* big = ws + 2 * MD;
    float* qb  = big;
    float* kbuf= big + MD;
    float* vbuf= big + 2 * MD;
    float* f1  = big;

    const dim3 blk(256);
    const dim3 gProj(Dc / BNt,   Mc / BMt);   // N=1024
    const dim3 gFF1 (DFFc / BNt, Mc / BMt);   // N=4096
    const dim3 gHead(Vc / BNt,   Mc / BMt);   // N=32000 (250 tiles)
    const dim3 gAttn(Bc * Hc, Sc / 128);

    embed_k<<<(Mc * Dc + 255) / 256, blk, 0, stream>>>(x, tok_emb, pos_emb, h, Mc * Dc);

    for (int i = 0; i < Lc; ++i) {
        const size_t dd  = (size_t)i * Dc * Dc;
        const size_t dff = (size_t)i * Dc * DFFc;
        layernorm_k<<<Mc, blk, 0, stream>>>(h, ln1_g + i * Dc, ln1_b + i * Dc, y, Dc);
        gemm_bf16_wmma<<<gProj, blk, 0, stream>>>(y, Wq + dd, bq + i * Dc, nullptr, qb,   Mc, Dc, Dc, 0);
        gemm_bf16_wmma<<<gProj, blk, 0, stream>>>(y, Wk + dd, bk + i * Dc, nullptr, kbuf, Mc, Dc, Dc, 0);
        gemm_bf16_wmma<<<gProj, blk, 0, stream>>>(y, Wv + dd, bv + i * Dc, nullptr, vbuf, Mc, Dc, Dc, 0);
        attention_k<<<gAttn, blk, 0, stream>>>(qb, kbuf, vbuf, y);     // attn out -> y
        gemm_bf16_wmma<<<gProj, blk, 0, stream>>>(y, Wo + dd, bo + i * Dc, h, h, Mc, Dc, Dc, 1);
        layernorm_k<<<Mc, blk, 0, stream>>>(h, ln2_g + i * Dc, ln2_b + i * Dc, y, Dc);
        gemm_bf16_wmma<<<gFF1, blk, 0, stream>>>(y,  W1 + dff, b1 + i * DFFc, nullptr, f1, Mc, DFFc, Dc, 2);
        gemm_bf16_wmma<<<gProj, blk, 0, stream>>>(f1, W2 + dff, b2 + i * Dc,  h,       h,  Mc, Dc, DFFc, 1);
    }

    layernorm_k<<<Mc, blk, 0, stream>>>(h, lnf_g, lnf_b, y, Dc);
    gemm_bf16_wmma<<<gHead, blk, 0, stream>>>(y, Wh, bh, nullptr, out, Mc, Vc, Dc, 0);
}
